// HSL_Layer_Part2_34368328303052
// MI455X (gfx1250) — matrix operand order repaired and checked by reading, not compiled, not run on hardware
//
#include <hip/hip_runtime.h>
#include <math.h>

typedef _Float16 v8h  __attribute__((ext_vector_type(8)));
typedef _Float16 v16h __attribute__((ext_vector_type(16)));
typedef float    v8f  __attribute__((ext_vector_type(8)));

#define N_CODES 4096
#define N_EDGES 8192
#define EMB     128
#define N_C     4
#define NNZ     65536
#define KDIM    (N_C * EMB)          /* 512 */
#define TOPK    6553                 /* int(0.1 * 65536) */
#define S_ELEMS ((size_t)N_CODES * N_EDGES)  /* 33554432 */
#define MASKV   (-1e9f)

// ---------- helpers ----------
__device__ __forceinline__ float waveReduceSum(float x) {
  #pragma unroll
  for (int off = 16; off > 0; off >>= 1) x += __shfl_xor(x, off, 32);
  return x;
}

__device__ __forceinline__ unsigned orderF(float f) {
  unsigned u = __float_as_uint(f);
  return (u & 0x80000000u) ? ~u : (u | 0x80000000u);
}

// Threefry-2x32, 20 rounds, key = (0, 42)  (jax.random.key(42))
__device__ __forceinline__ void tf2x32(unsigned& x0, unsigned& x1) {
  const unsigned ks0 = 0u, ks1 = 42u, ks2 = 0x1BD11BDAu ^ ks0 ^ ks1;
  const int RA[4] = {13, 15, 26, 6};
  const int RB[4] = {17, 29, 16, 24};
  x0 += ks0; x1 += ks1;
  #pragma unroll
  for (int r = 0; r < 4; ++r) { x0 += x1; x1 = (x1 << RA[r]) | (x1 >> (32 - RA[r])); x1 ^= x0; }
  x0 += ks1; x1 += ks2 + 1u;
  #pragma unroll
  for (int r = 0; r < 4; ++r) { x0 += x1; x1 = (x1 << RB[r]) | (x1 >> (32 - RB[r])); x1 ^= x0; }
  x0 += ks2; x1 += ks0 + 2u;
  #pragma unroll
  for (int r = 0; r < 4; ++r) { x0 += x1; x1 = (x1 << RA[r]) | (x1 >> (32 - RA[r])); x1 ^= x0; }
  x0 += ks0; x1 += ks1 + 3u;
  #pragma unroll
  for (int r = 0; r < 4; ++r) { x0 += x1; x1 = (x1 << RB[r]) | (x1 >> (32 - RB[r])); x1 ^= x0; }
  x0 += ks1; x1 += ks2 + 4u;
  #pragma unroll
  for (int r = 0; r < 4; ++r) { x0 += x1; x1 = (x1 << RA[r]) | (x1 >> (32 - RA[r])); x1 ^= x0; }
  x0 += ks2; x1 += ks0 + 5u;
}

// ---------- stage 1: scatter-mean accumulation ----------
__global__ __launch_bounds__(256)
void scatter_kernel(const float* __restrict__ X, const int* __restrict__ V,
                    const int* __restrict__ E, float* __restrict__ sums,
                    float* __restrict__ cnt) {
  int t = blockIdx.x * blockDim.x + threadIdx.x;
  if (t >= NNZ * EMB) return;
  int i = t >> 7, d = t & 127;
  int v = V[i], e = E[i];
  atomicAdd(&sums[(size_t)e * EMB + d], X[(size_t)v * EMB + d]);
  if (d == 0) atomicAdd(&cnt[e], 1.0f);
}

// ---------- stage 2: normalized f16 feature matrices ----------
__global__ __launch_bounds__(256)
void node_feat_kernel(const float* __restrict__ X, const float* __restrict__ W,
                      _Float16* __restrict__ out) {
  int wid  = blockIdx.x * (blockDim.x >> 5) + (threadIdx.x >> 5);
  if (wid >= N_CODES * N_C) return;
  int n = wid >> 2, h = wid & 3;
  int lane = threadIdx.x & 31;
  float v[4], ss = 0.0f;
  #pragma unroll
  for (int j = 0; j < 4; ++j) {
    int d = lane + j * 32;
    v[j] = X[(size_t)n * EMB + d] * W[(size_t)h * EMB + d];
    ss += v[j] * v[j];
  }
  ss = waveReduceSum(ss);
  float inv = 1.0f / fmaxf(sqrtf(ss), 1e-6f);
  #pragma unroll
  for (int j = 0; j < 4; ++j) {
    int d = lane + j * 32;
    out[(size_t)n * KDIM + h * EMB + d] = (_Float16)(v[j] * inv);
  }
}

__global__ __launch_bounds__(256)
void edge_feat_kernel(const float* __restrict__ sums, const float* __restrict__ cnt,
                      const float* __restrict__ W, _Float16* __restrict__ out) {
  int wid  = blockIdx.x * (blockDim.x >> 5) + (threadIdx.x >> 5);
  if (wid >= N_EDGES * N_C) return;
  int m = wid >> 2, h = wid & 3;
  int lane = threadIdx.x & 31;
  float invc = 1.0f / fmaxf(cnt[m], 1.0f);
  float v[4], ss = 0.0f;
  #pragma unroll
  for (int j = 0; j < 4; ++j) {
    int d = lane + j * 32;
    v[j] = (sums[(size_t)m * EMB + d] * invc) * W[(size_t)h * EMB + d];
    ss += v[j] * v[j];
  }
  ss = waveReduceSum(ss);
  float inv = 1.0f / fmaxf(sqrtf(ss), 1e-6f);
  #pragma unroll
  for (int j = 0; j < 4; ++j) {
    int d = lane + j * 32;
    out[(size_t)m * KDIM + h * EMB + d] = (_Float16)(v[j] * inv);
  }
}

// ---------- stage 3: WMMA GEMM  S = 0.25 * A(4096x512) * B(8192x512)^T ----------
// block = 4 waves (2x2), each wave computes a 32(M) x 64(N) tile
__global__ __launch_bounds__(128)
void gemm_wmma_kernel(const _Float16* __restrict__ A, const _Float16* __restrict__ B,
                      float* __restrict__ S) {
  const int lane  = threadIdx.x & 31;
  const int wave  = threadIdx.x >> 5;
  const int waveM = wave >> 1, waveN = wave & 1;
  const int m0 = blockIdx.y * 64 + waveM * 32;
  const int n0 = blockIdx.x * 128 + waveN * 64;
  const int r  = lane & 15;
  const int hi = lane >> 4;

  v8f acc[2][4];
  #pragma unroll
  for (int t = 0; t < 2; ++t)
    #pragma unroll
    for (int s = 0; s < 4; ++s) acc[t][s] = (v8f){0,0,0,0,0,0,0,0};

  const _Float16* arow[2];
  arow[0] = A + (size_t)(m0 + r) * KDIM;
  arow[1] = A + (size_t)(m0 + 16 + r) * KDIM;
  const _Float16* brow[4];
  #pragma unroll
  for (int s = 0; s < 4; ++s) brow[s] = B + (size_t)(n0 + s * 16 + r) * KDIM;

  for (int kb = 0; kb < KDIM; kb += 32) {
    union { v16h v; v8h h[2]; } a[2];
    #pragma unroll
    for (int t = 0; t < 2; ++t) {
      a[t].h[0] = *(const v8h*)(arow[t] + kb + hi * 8);        // K = hi*8 .. hi*8+7
      a[t].h[1] = *(const v8h*)(arow[t] + kb + 16 + hi * 8);   // K = 16+hi*8 ..
    }
    union { v16h v; v8h h[2]; } b[4];
    #pragma unroll
    for (int s = 0; s < 4; ++s) {
      b[s].h[0] = *(const v8h*)(brow[s] + kb + hi * 8);        // K = hi*8 .. hi*8+7
      b[s].h[1] = *(const v8h*)(brow[s] + kb + 16 + hi * 8);   // K = 16+hi*8 ..
    }
    #pragma unroll
    for (int t = 0; t < 2; ++t)
      #pragma unroll
      for (int s = 0; s < 4; ++s)
        acc[t][s] = __builtin_amdgcn_wmma_f32_16x16x32_f16(
            false, a[t].v, false, b[s].v, (short)0, acc[t][s], false, false);
  }

  #pragma unroll
  for (int t = 0; t < 2; ++t)
    #pragma unroll
    for (int s = 0; s < 4; ++s)
      #pragma unroll
      for (int v = 0; v < 8; ++v) {
        int mm = m0 + t * 16 + hi * 8 + v;
        int nn = n0 + s * 16 + r;
        S[(size_t)mm * N_EDGES + nn] = 0.25f * acc[t][s][v];  // mean over 4 heads
      }
}

// ---------- stage 4: mask existing incidences ----------
__global__ __launch_bounds__(256)
void mask_kernel(const int* __restrict__ V, const int* __restrict__ E,
                 float* __restrict__ S) {
  int i = blockIdx.x * blockDim.x + threadIdx.x;
  if (i >= NNZ) return;
  S[(size_t)V[i] * N_EDGES + E[i]] = MASKV;
}

// ---------- stage 5: radix-histogram top-k threshold ----------
__global__ __launch_bounds__(256)
void hist_hi_kernel(const float* __restrict__ S, unsigned* __restrict__ hist) {
  size_t i = (size_t)blockIdx.x * blockDim.x + threadIdx.x;
  if (i >= S_ELEMS) return;
  atomicAdd(&hist[orderF(S[i]) >> 16], 1u);
}

__global__ __launch_bounds__(256)
void scan_hi_kernel(const unsigned* __restrict__ hist, unsigned* __restrict__ meta) {
  __shared__ unsigned csum[256];
  int t = threadIdx.x;
  int top = 65535 - t * 256;
  unsigned s = 0;
  for (int j = 0; j < 256; ++j) s += hist[top - j];
  csum[t] = s;
  __syncthreads();
  if (t == 0) {
    unsigned cum = 0, bsel = 0, kprime = TOPK;
    for (int c = 0; c < 256; ++c) {
      if (cum + csum[c] >= (unsigned)TOPK) {
        int tp = 65535 - c * 256;
        for (int j = 0; j < 256; ++j) {
          unsigned h = hist[tp - j];
          if (cum + h >= (unsigned)TOPK) { bsel = (unsigned)(tp - j); kprime = TOPK - cum; break; }
          cum += h;
        }
        break;
      }
      cum += csum[c];
    }
    meta[0] = bsel;
    meta[1] = kprime;
  }
}

__global__ __launch_bounds__(256)
void hist_lo_kernel(const float* __restrict__ S, const unsigned* __restrict__ meta,
                    unsigned* __restrict__ hist2) {
  size_t i = (size_t)blockIdx.x * blockDim.x + threadIdx.x;
  if (i >= S_ELEMS) return;
  unsigned b = meta[0];
  unsigned u = orderF(S[i]);
  if ((u >> 16) == b) atomicAdd(&hist2[u & 0xFFFFu], 1u);
}

__global__ __launch_bounds__(256)
void scan_lo_kernel(const unsigned* __restrict__ hist2, unsigned* __restrict__ meta) {
  __shared__ unsigned csum[256];
  int t = threadIdx.x;
  int top = 65535 - t * 256;
  unsigned s = 0;
  for (int j = 0; j < 256; ++j) s += hist2[top - j];
  csum[t] = s;
  __syncthreads();
  if (t == 0) {
    unsigned K = meta[1];
    unsigned cum = 0, lsel = 0;
    for (int c = 0; c < 256; ++c) {
      if (cum + csum[c] >= K) {
        int tp = 65535 - c * 256;
        for (int j = 0; j < 256; ++j) {
          unsigned h = hist2[tp - j];
          if (cum + h >= K) { lsel = (unsigned)(tp - j); break; }
          cum += h;
        }
        break;
      }
      cum += csum[c];
    }
    meta[2] = (meta[0] << 16) | lsel;
  }
}

// ---------- stage 6: fused enrich + RelaxedBernoulli mask ----------
__global__ __launch_bounds__(256)
void final_kernel(const float* __restrict__ S, const float* __restrict__ H,
                  const float* __restrict__ P, const unsigned* __restrict__ meta,
                  float* __restrict__ out) {
  size_t i = (size_t)blockIdx.x * blockDim.x + threadIdx.x;
  if (i >= S_ELEMS) return;
  unsigned thr = meta[2];
  unsigned u = orderF(S[i]);
  float enriched = H[i] + ((u >= thr) ? 1.0f : 0.0f);

  // jax.random.uniform(key(42), shape) via threefry2x32 with even-size count split
  const unsigned half = (unsigned)(S_ELEMS / 2);
  unsigned idx = (unsigned)i;
  unsigned j = (idx < half) ? idx : (idx - half);
  unsigned x0 = j, x1 = j + half;
  tf2x32(x0, x1);
  unsigned bits = (idx < half) ? x0 : x1;
  float f  = __uint_as_float((bits >> 9) | 0x3f800000u) - 1.0f;   // [0,1)
  float uu = 1e-6f + f * (1.0f - 2e-6f);

  float p = P[i];
  float logits = (logf(p) - log1pf(-p) + logf(uu) - log1pf(-uu)) * 2.0f; // /TEMP(0.5)
  float m = 1.0f / (1.0f + expf(-logits));
  out[i] = enriched * m;
}

// ---------- launcher ----------
extern "C" void kernel_launch(void* const* d_in, const int* in_sizes, int n_in,
                              void* d_out, int out_size, void* d_ws, size_t ws_size,
                              hipStream_t stream) {
  const float* X = (const float*)d_in[0];
  const float* H = (const float*)d_in[1];
  const int*   V = (const int*)d_in[2];
  const int*   E = (const int*)d_in[3];
  const float* P = (const float*)d_in[4];
  const float* W = (const float*)d_in[5];
  float* out = (float*)d_out;

  char* ws = (char*)d_ws;
  size_t off = 0;
  float*     S     = (float*)(ws + off);     off += S_ELEMS * sizeof(float);          // 128 MB
  float*     sums  = (float*)(ws + off);     off += (size_t)N_EDGES * EMB * 4;        // 4 MB
  float*     cnt   = (float*)(ws + off);     off += (size_t)N_EDGES * 4;              // 32 KB
  _Float16*  nf    = (_Float16*)(ws + off);  off += (size_t)N_CODES * KDIM * 2;       // 4 MB
  _Float16*  ef    = (_Float16*)(ws + off);  off += (size_t)N_EDGES * KDIM * 2;       // 8 MB
  unsigned*  hist1 = (unsigned*)(ws + off);  off += 65536u * 4;                       // 256 KB
  unsigned*  hist2 = (unsigned*)(ws + off);  off += 65536u * 4;                       // 256 KB
  unsigned*  meta  = (unsigned*)(ws + off);  off += 256;

  // zero the accumulators (sums+cnt contiguous; hist1+hist2 contiguous)
  hipMemsetAsync(sums, 0, (size_t)N_EDGES * EMB * 4 + (size_t)N_EDGES * 4, stream);
  hipMemsetAsync(hist1, 0, 2u * 65536u * 4, stream);

  // 1) scatter-mean accumulation
  scatter_kernel<<<(NNZ * EMB + 255) / 256, 256, 0, stream>>>(X, V, E, sums, cnt);

  // 2) normalized f16 features (one wave per (row, head))
  node_feat_kernel<<<(N_CODES * N_C + 7) / 8, 256, 0, stream>>>(X, W, nf);
  edge_feat_kernel<<<(N_EDGES * N_C + 7) / 8, 256, 0, stream>>>(sums, cnt, W, ef);

  // 3) WMMA GEMM -> S
  dim3 ggrid(N_EDGES / 128, N_CODES / 64);
  gemm_wmma_kernel<<<ggrid, 128, 0, stream>>>(nf, ef, S);

  // 4) mask existing incidences
  mask_kernel<<<(NNZ + 255) / 256, 256, 0, stream>>>(V, E, S);

  // 5) top-k threshold via two-level radix histogram
  int eb = (int)((S_ELEMS + 255) / 256);
  hist_hi_kernel<<<eb, 256, 0, stream>>>(S, hist1);
  scan_hi_kernel<<<1, 256, 0, stream>>>(hist1, meta);
  hist_lo_kernel<<<eb, 256, 0, stream>>>(S, meta, hist2);
  scan_lo_kernel<<<1, 256, 0, stream>>>(hist2, meta);

  // 6) fused enrich + RelaxedBernoulli sample
  final_kernel<<<eb, 256, 0, stream>>>(S, H, P, meta, out);
}
